// SCSPA_25572235280931
// MI455X (gfx1250) — compile-verified, matrix operand order checked by reading, not run on hardware
//
#include <hip/hip_runtime.h>
#include <hip/hip_bf16.h>

// ---------------------------------------------------------------------------
// Problem constants (from reference): B=16, C=256, H=W=64
//   NPIX = B*H*W = 65536 ; per-image pixels = 4096 ; hid_ca=64 ; hid_pa=32
// ---------------------------------------------------------------------------

typedef __attribute__((ext_vector_type(16))) __bf16        v16bf;
typedef __attribute__((ext_vector_type(8)))  float         v8f;
typedef __attribute__((ext_vector_type(4)))  unsigned int  u32x4;

union BF16x16 { u32x4 q[2]; v16bf v; };

// ---- CDNA5 async global->LDS copy (ASYNCcnt-tracked), 16B per lane ---------
__device__ __forceinline__ void async_copy16(unsigned ldsaddr, const void* gaddr) {
    asm volatile("global_load_async_to_lds_b128 %0, %1, off"
                 :
                 : "v"(ldsaddr), "v"((unsigned long long)(uintptr_t)gaddr)
                 : "memory");
}
__device__ __forceinline__ void wait_async0() {
    asm volatile("s_wait_asynccnt 0x0" ::: "memory");
}

// ---------------------------------------------------------------------------
// K0a: pack fuse_w (256x256, row = out-ch, col = in-ch) into B-fragment order
//   dst[((ks*16+nt)*32+lane)*16 + i] = bf16( w[n][k] )
//   n = nt*16 + (lane&15); k = ks*32 + kk
//   kk: lane<16 -> {0..7,16..23}; lane>=16 -> {8..15,24..31}
// ---------------------------------------------------------------------------
__global__ void k_prep_fuse(const float* __restrict__ fw, __bf16* __restrict__ dst) {
    int idx  = blockIdx.x * 256 + threadIdx.x;       // 65536 elements
    int i    = idx & 15;
    int lane = (idx >> 4) & 31;
    int nt   = (idx >> 9) & 15;
    int ks   = idx >> 13;
    int n    = nt * 16 + (lane & 15);
    int kk   = (lane < 16) ? ((i < 8) ? i : i + 8) : ((i < 8) ? i + 8 : i + 16);
    dst[idx] = (__bf16)fw[n * 256 + ks * 32 + kk];
}

// K0b: pack pa_w1 (32x256) likewise (2 n-tiles, 8 k-steps -> 16384 elements)
__global__ void k_prep_pa(const float* __restrict__ pw1, __bf16* __restrict__ dst) {
    int idx  = blockIdx.x * 256 + threadIdx.x;       // 16384 elements
    int i    = idx & 15;
    int lane = (idx >> 4) & 31;
    int nt   = (idx >> 9) & 1;
    int ks   = idx >> 10;
    int n    = nt * 16 + (lane & 15);
    int kk   = (lane < 16) ? ((i < 8) ? i : i + 8) : ((i < 8) ? i + 8 : i + 16);
    dst[idx] = (__bf16)pw1[n * 256 + ks * 32 + kk];
}

// ---------------------------------------------------------------------------
// K1: per-(b,c) mean and max over the 64x64 image (one block per channel image)
// ---------------------------------------------------------------------------
__global__ __launch_bounds__(256) void k_pool(const float* __restrict__ x,
                                              float* __restrict__ avg,
                                              float* __restrict__ mx) {
    int bc = blockIdx.x;                              // 0..4095
    int t  = threadIdx.x;
    const float* p = x + (size_t)bc * 4096;
    float s = 0.f, m = -__builtin_inff();
    for (int i = t; i < 4096; i += 256) { float v = p[i]; s += v; m = fmaxf(m, v); }
    __shared__ float ss[256], sm[256];
    ss[t] = s; sm[t] = m;
    __syncthreads();
    for (int st = 128; st > 0; st >>= 1) {
        if (t < st) { ss[t] += ss[t + st]; sm[t] = fmaxf(sm[t], sm[t + st]); }
        __syncthreads();
    }
    if (t == 0) { avg[bc] = ss[0] * (1.f / 4096.f); mx[bc] = sm[0]; }
}

// ---------------------------------------------------------------------------
// K2: SE MLP for avg & max paths; gate = clip(sig(mlp(avg)) + sig(mlp(max)),0,1)
//     one block per batch element (16 blocks, 256 threads)
// ---------------------------------------------------------------------------
__global__ __launch_bounds__(256) void k_camlp(const float* __restrict__ avg,
                                               const float* __restrict__ mx,
                                               const float* __restrict__ w1,
                                               const float* __restrict__ b1,
                                               const float* __restrict__ w2,
                                               const float* __restrict__ b2,
                                               float* __restrict__ gate) {
    int b = blockIdx.x, t = threadIdx.x;
    __shared__ float pa_[256], pm_[256], ha[64], hm[64];
    pa_[t] = avg[b * 256 + t];
    pm_[t] = mx[b * 256 + t];
    __syncthreads();
    if (t < 64) {
        float aA = b1[t], aM = b1[t];
        for (int c = 0; c < 256; ++c) {
            float w = w1[t * 256 + c];
            aA += w * pa_[c]; aM += w * pm_[c];
        }
        ha[t] = fmaxf(aA, 0.f); hm[t] = fmaxf(aM, 0.f);
    }
    __syncthreads();
    float aA = b2[t], aM = b2[t];
    for (int h = 0; h < 64; ++h) {
        float w = w2[t * 64 + h];
        aA += w * ha[h]; aM += w * hm[h];
    }
    float g = 1.f / (1.f + __expf(-aA)) + 1.f / (1.f + __expf(-aM));
    gate[b * 256 + t] = fminf(fmaxf(g, 0.f), 1.f);
}

// ---------------------------------------------------------------------------
// K3: out_ca (= x * gate) packed as bf16 in [pixel][channel] layout for PA GEMM
// ---------------------------------------------------------------------------
__global__ __launch_bounds__(256) void k_pack_oca(const float* __restrict__ x,
                                                  const float* __restrict__ gate,
                                                  __bf16* __restrict__ oca) {
    size_t i = (size_t)blockIdx.x * 256 + threadIdx.x;   // 16.7M elements, NCHW order
    int    hw = (int)(i & 4095);
    size_t tt = i >> 12;                                  // b*256 + c
    int    c  = (int)(tt & 255);
    int    b  = (int)(tt >> 8);
    float  v  = x[i] * gate[tt];
    oca[((size_t)(b * 4096 + hw)) * 256 + c] = (__bf16)v;
}

// ---------------------------------------------------------------------------
// K4: depthwise conv stack entirely in LDS. One block per (b,c) channel image.
//     x0 = dw5x5(out_ca); s = x0 + dw7x1(dw1x7(x0)) + dw11x1(dw1x11(x0))
//                              + dw21x1(dw1x21(x0));   s -> bf16 [pix][c]
// ---------------------------------------------------------------------------
__global__ __launch_bounds__(256) void k_dwconv(
    const float* __restrict__ x, const float* __restrict__ gate,
    const float* __restrict__ d55,
    const float* __restrict__ wh7,  const float* __restrict__ wv7,
    const float* __restrict__ wh11, const float* __restrict__ wv11,
    const float* __restrict__ wh21, const float* __restrict__ wv21,
    __bf16* __restrict__ sbuf) {
    __shared__ float A_[4096];   // 16 KB
    __shared__ float B_[4096];   // 16 KB
    __shared__ float wS[103];    // 25 + 2*(7+11+21)
    int bc = blockIdx.x;
    int c  = bc & 255;
    int b  = bc >> 8;
    int t  = threadIdx.x;

    if      (t < 25)  wS[t] = d55 [c * 25 + t];
    else if (t < 32)  wS[t] = wh7 [c * 7  + (t - 25)];
    else if (t < 39)  wS[t] = wv7 [c * 7  + (t - 32)];
    else if (t < 50)  wS[t] = wh11[c * 11 + (t - 39)];
    else if (t < 61)  wS[t] = wv11[c * 11 + (t - 50)];
    else if (t < 82)  wS[t] = wh21[c * 21 + (t - 61)];
    else if (t < 103) wS[t] = wv21[c * 21 + (t - 82)];

    const float g = gate[bc];
    const float* img = x + (size_t)bc * 4096;
    for (int i = t; i < 4096; i += 256) A_[i] = img[i] * g;
    __syncthreads();

    int row  = t >> 2;
    int col0 = (t & 3) << 4;        // each thread owns 16 contiguous pixels of one row
    float sum[16];

    // x0 = 5x5 depthwise with zero padding
    #pragma unroll 4
    for (int j = 0; j < 16; ++j) {
        int col = col0 + j;
        float a = 0.f;
        for (int ky = 0; ky < 5; ++ky) {
            int rr = row + ky - 2;
            if ((unsigned)rr < 64u)
                for (int kx = 0; kx < 5; ++kx) {
                    int cc = col + kx - 2;
                    if ((unsigned)cc < 64u) a += A_[rr * 64 + cc] * wS[ky * 5 + kx];
                }
        }
        sum[j] = a;
        B_[row * 64 + col] = a;     // B_ = x0
    }
    __syncthreads();

    const int Ks[3]    = {7, 11, 21};
    const int whOff[3] = {25, 39, 61};
    const int wvOff[3] = {32, 50, 82};
    for (int p = 0; p < 3; ++p) {
        int K = Ks[p], R = K >> 1;
        const float* wh = &wS[whOff[p]];
        const float* wv = &wS[wvOff[p]];
        // horizontal pass: B_ (x0) -> A_
        for (int j = 0; j < 16; ++j) {
            int col = col0 + j;
            float a = 0.f;
            for (int k = 0; k < K; ++k) {
                int cc = col + k - R;
                if ((unsigned)cc < 64u) a += B_[row * 64 + cc] * wh[k];
            }
            A_[row * 64 + col] = a;
        }
        __syncthreads();
        // vertical pass: A_ -> per-thread accumulator
        for (int j = 0; j < 16; ++j) {
            int col = col0 + j;
            float a = 0.f;
            for (int k = 0; k < K; ++k) {
                int rr = row + k - R;
                if ((unsigned)rr < 64u) a += A_[rr * 64 + col] * wv[k];
            }
            sum[j] += a;
        }
        __syncthreads();
    }

    // write s as bf16 [pix][channel]
    #pragma unroll 4
    for (int j = 0; j < 16; ++j) {
        int hw = row * 64 + col0 + j;
        sbuf[((size_t)(b * 4096 + hw)) * 256 + c] = (__bf16)sum[j];
    }
}

// ---------------------------------------------------------------------------
// K5: fuse 1x1 conv as bf16 WMMA GEMM: (65536 x 256) = (65536 x 256)*(256 x 256)
//     block = 8 waves: 4 M-tiles x 2 N-halves; each wave: 1 Mtile x 8 Ntiles.
//     Weights double-buffered in LDS, filled with async global->LDS DMA
//     (ASYNCcnt) while the current k-step's WMMAs run.
//     Epilogue: sigmoid, store bf16 [pix][channel]; fuse_b preloaded into acc.
// ---------------------------------------------------------------------------
__global__ __launch_bounds__(256) void k_gemm_fuse(const __bf16* __restrict__ sbuf,
                                                   const __bf16* __restrict__ wB,
                                                   const float*  __restrict__ fuse_b,
                                                   __bf16* __restrict__ sa) {
    __shared__ __attribute__((aligned(32))) __bf16 lds[2][8192];   // 2 x 16 KB
    int t    = threadIdx.x;
    int lane = t & 31;
    int w    = t >> 5;          // wave id 0..7
    int mi   = w & 3;           // M-tile within block
    int nh   = w >> 2;          // N half (0: n 0..127, 1: n 128..255)
    int Mbase  = blockIdx.x * 64 + mi * 16;
    int nBase0 = nh * 128;

    v8f acc[8];
    #pragma unroll
    for (int nt = 0; nt < 8; ++nt) {
        float bias = fuse_b[nBase0 + nt * 16 + (lane & 15)];
        #pragma unroll
        for (int r = 0; r < 8; ++r) acc[nt][r] = bias;
    }

    int rowA = Mbase + (lane & 15);
    const __bf16* arow = sbuf + (size_t)rowA * 256 + ((lane >> 4) << 3);
    const char*   wby  = (const char*)wB;
    unsigned ldsA0 = (unsigned)(uintptr_t)&lds[0][0] + (unsigned)t * 16u;
    unsigned ldsA1 = (unsigned)(uintptr_t)&lds[1][0] + (unsigned)t * 16u;

    // prefill buffer 0 with k-step 0 weights (async DMA, each thread 64B)
    {
        const char* ga = wby + t * 16;
        #pragma unroll
        for (int i = 0; i < 4; ++i) async_copy16(ldsA0 + i * 4096u, ga + i * 4096);
        wait_async0();
        __syncthreads();
    }

    for (int ks = 0; ks < 8; ++ks) {
        int cur = ks & 1;
        // kick off async fill of the other buffer with next k-step's weights
        if (ks < 7) {
            unsigned la  = cur ? ldsA0 : ldsA1;
            const char* ga = wby + (size_t)(ks + 1) * 16384 + t * 16;
            #pragma unroll
            for (int i = 0; i < 4; ++i) async_copy16(la + i * 4096u, ga + i * 4096);
        }

        BF16x16 af;
        af.q[0] = *(const u32x4*)(arow + ks * 32);        // K lo chunk
        af.q[1] = *(const u32x4*)(arow + ks * 32 + 16);   // K hi chunk
        if (ks < 7) __builtin_prefetch(arow + ks * 32 + 32, 0, 0);

        #pragma unroll
        for (int nt = 0; nt < 8; ++nt) {
            const v16bf* bp =
                (const v16bf*)(&lds[cur][(size_t)((nh * 8 + nt) * 32 + lane) * 16]);
            acc[nt] = __builtin_amdgcn_wmma_f32_16x16x32_bf16(
                false, af.v, false, *bp, (short)0, acc[nt], false, false);
        }

        wait_async0();       // this wave's DMA for next buffer has landed in LDS
        __syncthreads();     // everyone's portion landed; safe to switch buffers
    }

    int mAdd = (lane >> 4) << 3;
    int n    = lane & 15;
    #pragma unroll
    for (int nt = 0; nt < 8; ++nt) {
        int nIdx = nBase0 + nt * 16 + n;
        #pragma unroll
        for (int r = 0; r < 8; ++r) {
            float v  = acc[nt][r];
            float sg = 1.f / (1.f + __expf(-v));
            int pix  = Mbase + r + mAdd;
            sa[(size_t)pix * 256 + nIdx] = (__bf16)sg;
        }
    }
}

// ---------------------------------------------------------------------------
// K6: pixel-attention: ph = relu(oca @ pa_w1^T + b1) (WMMA, N=32),
//     pa = sigmoid(ph . pa_w2 + b2) folded into the epilogue with wave32
//     half-wave shuffle reductions. Weights async-DMA'd to LDS once.
// ---------------------------------------------------------------------------
__global__ __launch_bounds__(256) void k_gemm_pa(const __bf16* __restrict__ oca,
                                                 const __bf16* __restrict__ wB,
                                                 const float*  __restrict__ pa_b1,
                                                 const float*  __restrict__ pa_w2,
                                                 const float*  __restrict__ pa_b2,
                                                 float* __restrict__ pa) {
    __shared__ __attribute__((aligned(32))) __bf16 lds[8192];   // all packed pa_w1
    int t    = threadIdx.x;
    int lane = t & 31;
    int w    = t >> 5;
    {
        unsigned la = (unsigned)(uintptr_t)&lds[0] + (unsigned)t * 16u;
        const char* ga = (const char*)wB + t * 16;
        #pragma unroll
        for (int i = 0; i < 4; ++i) async_copy16(la + i * 4096u, ga + i * 4096);
        wait_async0();
    }
    __syncthreads();

    int Mbase = blockIdx.x * 128 + w * 16;
    float b1a = pa_b1[lane & 15];
    float b1b = pa_b1[16 + (lane & 15)];
    v8f acc0, acc1;
    #pragma unroll
    for (int r = 0; r < 8; ++r) { acc0[r] = b1a; acc1[r] = b1b; }

    int rowA = Mbase + (lane & 15);
    const __bf16* arow = oca + (size_t)rowA * 256 + ((lane >> 4) << 3);

    #pragma unroll
    for (int ks = 0; ks < 8; ++ks) {
        BF16x16 af;
        af.q[0] = *(const u32x4*)(arow + ks * 32);
        af.q[1] = *(const u32x4*)(arow + ks * 32 + 16);
        const v16bf* b0 = (const v16bf*)(&lds[(size_t)((ks * 2 + 0) * 32 + lane) * 16]);
        const v16bf* b1 = (const v16bf*)(&lds[(size_t)((ks * 2 + 1) * 32 + lane) * 16]);
        acc0 = __builtin_amdgcn_wmma_f32_16x16x32_bf16(false, af.v, false, *b0,
                                                       (short)0, acc0, false, false);
        acc1 = __builtin_amdgcn_wmma_f32_16x16x32_bf16(false, af.v, false, *b1,
                                                       (short)0, acc1, false, false);
    }

    float w2a = pa_w2[lane & 15];
    float w2b = pa_w2[16 + (lane & 15)];
    float bb2 = pa_b2[0];
    #pragma unroll
    for (int r = 0; r < 8; ++r) {
        float v = fmaxf(acc0[r], 0.f) * w2a + fmaxf(acc1[r], 0.f) * w2b;
        // reduce within each 16-lane half (row M=.. lives in lanes 0-15 / 16-31)
        v += __shfl_xor(v, 1);
        v += __shfl_xor(v, 2);
        v += __shfl_xor(v, 4);
        v += __shfl_xor(v, 8);
        if ((lane & 15) == 0) {
            int pix = Mbase + r + ((lane >> 4) << 3);
            pa[pix] = 1.f / (1.f + __expf(-(v + bb2)));
        }
    }
}

// ---------------------------------------------------------------------------
// K7: final elementwise, NCHW-coalesced:
//     out = out_ca * (1 + gamma*(sa*pa - 1)), out_ca recomputed as x*gate (fp32)
// ---------------------------------------------------------------------------
__global__ __launch_bounds__(256) void k_final(const float* __restrict__ x,
                                               const float* __restrict__ gate,
                                               const __bf16* __restrict__ sa,
                                               const float* __restrict__ pa,
                                               const float* __restrict__ gamma,
                                               float* __restrict__ out) {
    size_t i  = (size_t)blockIdx.x * 256 + threadIdx.x;
    int    hw = (int)(i & 4095);
    size_t tt = i >> 12;                 // b*256 + c
    int    c  = (int)(tt & 255);
    int    b  = (int)(tt >> 8);
    float  oc = x[i] * gate[tt];
    int    pix = b * 4096 + hw;
    float  sv = (float)sa[(size_t)pix * 256 + c];
    float  pv = pa[pix];
    float  g  = gamma[0];
    out[i] = oc * (1.f + g * (sv * pv - 1.f));
}

// ---------------------------------------------------------------------------
// Launch
// ---------------------------------------------------------------------------
extern "C" void kernel_launch(void* const* d_in, const int* in_sizes, int n_in,
                              void* d_out, int out_size, void* d_ws, size_t ws_size,
                              hipStream_t stream) {
    (void)in_sizes; (void)n_in; (void)out_size; (void)ws_size;
    const float* x      = (const float*)d_in[0];
    const float* ca_w1  = (const float*)d_in[1];
    const float* ca_b1  = (const float*)d_in[2];
    const float* ca_w2  = (const float*)d_in[3];
    const float* ca_b2  = (const float*)d_in[4];
    const float* d5_5   = (const float*)d_in[5];
    const float* d1_7   = (const float*)d_in[6];
    const float* d7_1   = (const float*)d_in[7];
    const float* d1_11  = (const float*)d_in[8];
    const float* d11_1  = (const float*)d_in[9];
    const float* d1_21  = (const float*)d_in[10];
    const float* d21_1  = (const float*)d_in[11];
    const float* fuse_w = (const float*)d_in[12];
    const float* fuse_b = (const float*)d_in[13];
    const float* pa_w1  = (const float*)d_in[14];
    const float* pa_b1  = (const float*)d_in[15];
    const float* pa_w2  = (const float*)d_in[16];
    const float* pa_b2  = (const float*)d_in[17];
    const float* gamma  = (const float*)d_in[18];
    float* out = (float*)d_out;

    char* ws = (char*)d_ws;
    float*  avg  = (float*)(ws + 0);                  //  16 KB
    float*  mx   = (float*)(ws + 16384);              //  16 KB
    float*  gate = (float*)(ws + 32768);              //  16 KB
    __bf16* wBf  = (__bf16*)(ws + 65536);             // 128 KB packed fuse_w
    __bf16* wBp  = (__bf16*)(ws + 196608);            //  32 KB packed pa_w1
    float*  paB  = (float*)(ws + 262144);             // 256 KB pa gate
    __bf16* oca  = (__bf16*)(ws + 524288);            //  32 MB out_ca bf16 [pix][c]
    __bf16* sbf  = (__bf16*)(ws + 34078720ull);       //  32 MB s bf16 [pix][c]
    __bf16* saB  = (__bf16*)(ws + 67633152ull);       //  32 MB sa bf16 [pix][c]

    k_prep_fuse<<<256, 256, 0, stream>>>(fuse_w, wBf);
    k_prep_pa  <<<64, 256, 0, stream>>>(pa_w1, wBp);
    k_pool     <<<4096, 256, 0, stream>>>(x, avg, mx);
    k_camlp    <<<16, 256, 0, stream>>>(avg, mx, ca_w1, ca_b1, ca_w2, ca_b2, gate);
    k_pack_oca <<<65536, 256, 0, stream>>>(x, gate, oca);
    k_dwconv   <<<4096, 256, 0, stream>>>(x, gate, d5_5, d1_7, d7_1, d1_11, d11_1,
                                          d1_21, d21_1, sbf);
    k_gemm_fuse<<<1024, 256, 0, stream>>>(sbf, wBf, fuse_b, saB);
    k_gemm_pa  <<<512, 256, 0, stream>>>(oca, wBp, pa_b1, pa_w2, pa_b2, paB);
    k_final    <<<65536, 256, 0, stream>>>(x, gate, saB, paB, gamma, out);
}
